// ScaledDotProductAttention_74620761800949
// MI455X (gfx1250) — matrix operand order repaired and checked by reading, not compiled, not run on hardware
//
#include <hip/hip_runtime.h>
#include <hip/hip_bf16.h>

typedef __attribute__((ext_vector_type(16))) _Float16 v16h;
typedef __attribute__((ext_vector_type(8)))  _Float16 v8h;
typedef __attribute__((ext_vector_type(4)))  _Float16 v4h;
typedef __attribute__((ext_vector_type(8)))  float    v8f;
typedef __attribute__((ext_vector_type(4)))  unsigned int u32x4;
typedef __attribute__((ext_vector_type(8)))  int i32x8;
typedef __attribute__((ext_vector_type(4)))  int i32x4;

#define S_LEN   2048
#define DKK     64
#define BHN     32                 // B*H
#define BC      64                 // keys per staged chunk
#define QW      16                 // query rows per wave
#define WAVES   8
#define QB      (QW * WAVES)       // 128 query rows per block
#define LSTR    72                 // LDS row stride in halves (64 + 8 pad)
#define NEG_BIG (-3.0e38f)         // finite stand-in for -inf in softmax math

union V16U { v16h v; v8h h[2]; };

__device__ __forceinline__ v8h cvt8s(const float* p, float s) {
  v8h r;
#pragma unroll
  for (int i = 0; i < 8; ++i) r[i] = (_Float16)(p[i] * s);
  return r;
}

__device__ __forceinline__ v8f wmma_f16(v16h a, v16h b, v8f c) {
  return __builtin_amdgcn_wmma_f32_16x16x32_f16(false, a, false, b, (short)0, c,
                                                false, false);
}

// ---------------------------------------------------------------------------
// TDM: 2D-tile tensor_load_to_lds, D# per CDNA5 ISA §8.
// pad: after every 32 DWORDs (one 64-half row) insert 4 DWORDs -> 72-half rows.
// ---------------------------------------------------------------------------
__device__ __forceinline__ void tdm_load_2d(unsigned int lds_addr,
                                            unsigned long long gaddr,
                                            unsigned int tensor_d0,
                                            unsigned int tensor_d1,
                                            unsigned long long stride0,
                                            unsigned int tile_d0,
                                            unsigned int tile_d1) {
  const unsigned int padw =
      (1u << 20) | (4u << 22) | (3u << 25);     // pad_enable, 32dw interval, 4dw pad
  u32x4 g0;
  g0[0] = 1u;                                   // count=1 (valid), user mode
  g0[1] = lds_addr;
  g0[2] = (unsigned int)(gaddr & 0xffffffffull);
  g0[3] = (unsigned int)((gaddr >> 32) & 0x1ffffffull) | (2u << 30);  // type=2
  i32x8 g1;
  g1[0] = (int)((1u << 16) | padw);             // data_size=1 (2B) + pad fields
  g1[1] = (int)(tensor_d0 << 16);
  g1[2] = (int)((tensor_d0 >> 16) | (tensor_d1 << 16));
  g1[3] = (int)((tensor_d1 >> 16) | (tile_d0 << 16));
  g1[4] = (int)(tile_d1 & 0xffffu);
  g1[5] = (int)(unsigned int)(stride0 & 0xffffffffull);
  g1[6] = (int)(unsigned int)((stride0 >> 32) & 0xffffull);
  g1[7] = 0;
  i32x4 z4 = {0, 0, 0, 0};
#if defined(__clang_major__) && (__clang_major__ >= 23)
  i32x8 z8 = {0, 0, 0, 0, 0, 0, 0, 0};
  __builtin_amdgcn_tensor_load_to_lds(g0, g1, z4, z4, z8, 0);
#else
  __builtin_amdgcn_tensor_load_to_lds(g0, g1, z4, z4, 0);
#endif
}

// ---------------------------------------------------------------------------
// Pre-pass: K -> f16 (same layout), V -> f16 transposed [bh][dv][key].
// ---------------------------------------------------------------------------
__global__ void cvt_kv(const float* __restrict__ K, const float* __restrict__ V,
                       _Float16* __restrict__ Kh, _Float16* __restrict__ Vt) {
  size_t i = (size_t)blockIdx.x * 256 + threadIdx.x;   // 0 .. 4M-1
  Kh[i] = (_Float16)K[i];
  size_t bh  = i >> 17;                // S*D = 131072
  size_t rem = i & 131071;
  size_t s   = rem >> 6;
  size_t d   = rem & 63;
  Vt[(bh << 17) + (d << 11) + s] = (_Float16)V[i];
}

// ---------------------------------------------------------------------------
// Flash attention, f16 WMMA, double-buffered TDM pipeline.
// ---------------------------------------------------------------------------
__launch_bounds__(256)
__global__ void fa_fwd_tdm(const float* __restrict__ Q,
                           const _Float16* __restrict__ Kh,
                           const _Float16* __restrict__ Vt,
                           float* __restrict__ O) {
  __shared__ _Float16 sKd[2 * BC  * LSTR];      // ping-pong [key][dk]
  __shared__ _Float16 sVd[2 * DKK * LSTR];      // ping-pong [dv][key]
  __shared__ _Float16 sP [WAVES * QW * LSTR];   // per-wave P tile

  const int tid  = threadIdx.x;
  const int wave = tid >> 5;
  const int lane = tid & 31;
  const int grp  = lane >> 4;
  const int ln16 = lane & 15;

  const int qblocks = S_LEN / QB;
  const int bh   = blockIdx.x / qblocks;
  const int qblk = blockIdx.x % qblocks;
  const int q0b  = qblk * QB;
  const int q0   = q0b + wave * QW;

  const size_t base = (size_t)bh * S_LEN * DKK;

  // Q tile -> f16 A-layout, softmax scale folded in.
  const float* qrow = Q + base + (size_t)(q0 + ln16) * DKK;
  V16U aq0, aq1;
  aq0.h[0] = cvt8s(qrow + grp * 8, 0.125f);
  aq0.h[1] = cvt8s(qrow + 16 + grp * 8, 0.125f);
  aq1.h[0] = cvt8s(qrow + 32 + grp * 8, 0.125f);
  aq1.h[1] = cvt8s(qrow + 48 + grp * 8, 0.125f);

  v8f oacc[4];
#pragma unroll
  for (int t = 0; t < 4; ++t) oacc[t] = (v8f){};
  float mrow[8], lrow[8];
#pragma unroll
  for (int r = 0; r < 8; ++r) { mrow[r] = -__builtin_inff(); lrow[r] = 0.0f; }

  const unsigned int ldsK = (unsigned int)(size_t)(&sKd[0]);
  const unsigned int ldsV = (unsigned int)(size_t)(&sVd[0]);
  const unsigned int kBufB = BC  * LSTR * 2;    // bytes per K buffer
  const unsigned int vBufB = DKK * LSTR * 2;    // bytes per V buffer

  const int nch = (q0b + QB) / BC;              // causal chunk count for block
  // ---- prologue: DMA chunk 0 into buffer 0 ----
  if (wave == 0) {
    tdm_load_2d(ldsK, (unsigned long long)(size_t)(Kh + base),
                DKK, S_LEN, DKK, DKK, BC);
    tdm_load_2d(ldsV, (unsigned long long)(size_t)(Vt + base),
                S_LEN, DKK, S_LEN, BC, DKK);
  }

  int p = 0;
  for (int i = 0; i < nch; ++i) {
    const int k0 = i * BC;
    if (wave == 0) {
      if (i + 1 < nch) {
        // prefetch next chunk into the other buffer, overlap with compute
        tdm_load_2d(ldsK + (p ^ 1) * kBufB,
                    (unsigned long long)(size_t)(Kh + base + (size_t)(k0 + BC) * DKK),
                    DKK, S_LEN, DKK, DKK, BC);
        tdm_load_2d(ldsV + (p ^ 1) * vBufB,
                    (unsigned long long)(size_t)(Vt + base + (k0 + BC)),
                    S_LEN, DKK, S_LEN, BC, DKK);
        __builtin_amdgcn_s_wait_tensorcnt(2);   // in-order: current buffer done
      } else {
        __builtin_amdgcn_s_wait_tensorcnt(0);
      }
    }
    __syncthreads();                            // publish current buffer

    if (k0 <= q0 + QW - 1) {                    // wave-uniform causal skip
      const _Float16* sK  = &sKd[p * (BC  * LSTR)];
      const _Float16* sVt = &sVd[p * (DKK * LSTR)];

      // ---- S = (Q K^T), scale pre-folded into Q ----
      v8f st[4];
#pragma unroll
      for (int t = 0; t < 4; ++t) {
        const _Float16* krow = &sK[(t * 16 + ln16) * LSTR];
        V16U b0, b1;
        b0.h[0] = *(const v8h*)(krow + grp * 8);
        b0.h[1] = *(const v8h*)(krow + 16 + grp * 8);
        b1.h[0] = *(const v8h*)(krow + 32 + grp * 8);
        b1.h[1] = *(const v8h*)(krow + 48 + grp * 8);
        v8f acc = (v8f){};
        acc = wmma_f16(aq0.v, b0.v, acc);
        acc = wmma_f16(aq1.v, b1.v, acc);
        st[t] = acc;
      }

      // ---- causal mask ----
      if ((k0 + BC - 1) > q0) {
#pragma unroll
        for (int t = 0; t < 4; ++t)
#pragma unroll
          for (int r = 0; r < 8; ++r) {
            int keyg = k0 + t * 16 + ln16;
            int qg   = q0 + r + 8 * grp;
            st[t][r] = (keyg > qg) ? -__builtin_inff() : st[t][r];
          }
      }

      // ---- online softmax; row-sum reduction deferred to epilogue ----
#pragma unroll
      for (int r = 0; r < 8; ++r) {
        float rmax = fmaxf(fmaxf(st[0][r], st[1][r]), fmaxf(st[2][r], st[3][r]));
#pragma unroll
        for (int off = 1; off < 16; off <<= 1)
          rmax = fmaxf(rmax, __shfl_xor(rmax, off, 32));
        float mo  = mrow[r];
        float mn  = fmaxf(mo, rmax);
        float mnc = fmaxf(mn, NEG_BIG);         // avoid -inf arithmetic
        float cf  = __expf(mo - mnc);           // 0 when mo==-inf (l,O are 0)
        float rs  = 0.0f;
#pragma unroll
        for (int t = 0; t < 4; ++t) {
          float pexp = __expf(st[t][r] - mnc);  // masked -> exp(-inf) = 0
          st[t][r] = pexp;
          rs += pexp;
        }
        lrow[r] = lrow[r] * cf + rs;            // per-lane partial sum
        mrow[r] = mn;
#pragma unroll
        for (int t = 0; t < 4; ++t) oacc[t][r] *= cf;
      }

      // ---- P: D-layout f32 -> A-layout f16 via per-wave LDS round-trip ----
      _Float16* sPw = &sP[wave * QW * LSTR];
#pragma unroll
      for (int t = 0; t < 4; ++t)
#pragma unroll
        for (int r = 0; r < 8; ++r)
          sPw[(r + 8 * grp) * LSTR + t * 16 + ln16] = (_Float16)st[t][r];
      __builtin_amdgcn_wave_barrier();          // LDS ops in-order within wave
      const _Float16* prow = &sPw[ln16 * LSTR];
      V16U ap0, ap1;
      ap0.h[0] = *(const v8h*)(prow + grp * 8);
      ap0.h[1] = *(const v8h*)(prow + 16 + grp * 8);
      ap1.h[0] = *(const v8h*)(prow + 32 + grp * 8);
      ap1.h[1] = *(const v8h*)(prow + 48 + grp * 8);

      // ---- O += P V ----
#pragma unroll
      for (int t = 0; t < 4; ++t) {
        const _Float16* vrow = &sVt[(t * 16 + ln16) * LSTR];
        V16U bv0, bv1;
        bv0.h[0] = *(const v8h*)(vrow + grp * 8);
        bv0.h[1] = *(const v8h*)(vrow + 16 + grp * 8);
        bv1.h[0] = *(const v8h*)(vrow + 32 + grp * 8);
        bv1.h[1] = *(const v8h*)(vrow + 48 + grp * 8);
        oacc[t] = wmma_f16(ap0.v, bv0.v, oacc[t]);
        oacc[t] = wmma_f16(ap1.v, bv1.v, oacc[t]);
      }
    }
    __syncthreads();                            // reads of buf[p] complete
    p ^= 1;
  }

  // ---- epilogue: finish l reduction, normalize, store fp32 ----
  float* obase = O + base;
#pragma unroll
  for (int r = 0; r < 8; ++r) {
    float ls = lrow[r];
#pragma unroll
    for (int off = 1; off < 16; off <<= 1)
      ls += __shfl_xor(ls, off, 32);
    float inv = 1.0f / ls;
    int m = q0 + r + 8 * grp;
    float* orow = obase + (size_t)m * DKK;
#pragma unroll
    for (int t = 0; t < 4; ++t)
      orow[t * 16 + ln16] = oacc[t][r] * inv;
  }
}

// ---------------------------------------------------------------------------
// Fallback: self-contained kernel with manual staging (no workspace needed).
// ---------------------------------------------------------------------------
__launch_bounds__(256)
__global__ void fa_fwd_manual(const float* __restrict__ Q,
                              const float* __restrict__ K,
                              const float* __restrict__ V,
                              float* __restrict__ O) {
  __shared__ _Float16 sK [BC  * LSTR];
  __shared__ _Float16 sVt[DKK * LSTR];
  __shared__ _Float16 sP [WAVES * QW * LSTR];

  const int tid  = threadIdx.x;
  const int wave = tid >> 5;
  const int lane = tid & 31;
  const int grp  = lane >> 4;
  const int ln16 = lane & 15;

  const int qblocks = S_LEN / QB;
  const int bh   = blockIdx.x / qblocks;
  const int qblk = blockIdx.x % qblocks;
  const int q0b  = qblk * QB;
  const int q0   = q0b + wave * QW;

  const size_t base = (size_t)bh * S_LEN * DKK;

  const float* qrow = Q + base + (size_t)(q0 + ln16) * DKK;
  V16U aq0, aq1;
  aq0.h[0] = cvt8s(qrow + grp * 8, 0.125f);
  aq0.h[1] = cvt8s(qrow + 16 + grp * 8, 0.125f);
  aq1.h[0] = cvt8s(qrow + 32 + grp * 8, 0.125f);
  aq1.h[1] = cvt8s(qrow + 48 + grp * 8, 0.125f);

  v8f oacc[4];
#pragma unroll
  for (int t = 0; t < 4; ++t) oacc[t] = (v8f){};
  float mrow[8], lrow[8];
#pragma unroll
  for (int r = 0; r < 8; ++r) { mrow[r] = -__builtin_inff(); lrow[r] = 0.0f; }

  const int kEnd = q0b + QB;
  for (int k0 = 0; k0 < kEnd; k0 += BC) {
    __syncthreads();
#pragma unroll
    for (int i = 0; i < 4; ++i) {
      int idx  = tid + i * 256;
      int row  = idx >> 4;
      int col4 = idx & 15;
      const float4 kv = *(const float4*)(K + base + (size_t)(k0 + row) * DKK + col4 * 4);
      v4h kh; kh[0] = (_Float16)kv.x; kh[1] = (_Float16)kv.y;
              kh[2] = (_Float16)kv.z; kh[3] = (_Float16)kv.w;
      *(v4h*)(&sK[row * LSTR + col4 * 4]) = kh;
      const float4 vv = *(const float4*)(V + base + (size_t)(k0 + row) * DKK + col4 * 4);
      sVt[(col4 * 4 + 0) * LSTR + row] = (_Float16)vv.x;
      sVt[(col4 * 4 + 1) * LSTR + row] = (_Float16)vv.y;
      sVt[(col4 * 4 + 2) * LSTR + row] = (_Float16)vv.z;
      sVt[(col4 * 4 + 3) * LSTR + row] = (_Float16)vv.w;
    }
    __syncthreads();

    if (k0 > q0 + QW - 1) continue;

    v8f st[4];
#pragma unroll
    for (int t = 0; t < 4; ++t) {
      const _Float16* krow = &sK[(t * 16 + ln16) * LSTR];
      V16U b0, b1;
      b0.h[0] = *(const v8h*)(krow + grp * 8);
      b0.h[1] = *(const v8h*)(krow + 16 + grp * 8);
      b1.h[0] = *(const v8h*)(krow + 32 + grp * 8);
      b1.h[1] = *(const v8h*)(krow + 48 + grp * 8);
      v8f acc = (v8f){};
      acc = wmma_f16(aq0.v, b0.v, acc);
      acc = wmma_f16(aq1.v, b1.v, acc);
      st[t] = acc;
    }

    if ((k0 + BC - 1) > q0) {
#pragma unroll
      for (int t = 0; t < 4; ++t)
#pragma unroll
        for (int r = 0; r < 8; ++r) {
          int keyg = k0 + t * 16 + ln16;
          int qg   = q0 + r + 8 * grp;
          st[t][r] = (keyg > qg) ? -__builtin_inff() : st[t][r];
        }
    }

#pragma unroll
    for (int r = 0; r < 8; ++r) {
      float rmax = fmaxf(fmaxf(st[0][r], st[1][r]), fmaxf(st[2][r], st[3][r]));
#pragma unroll
      for (int off = 1; off < 16; off <<= 1)
        rmax = fmaxf(rmax, __shfl_xor(rmax, off, 32));
      float mo  = mrow[r];
      float mn  = fmaxf(mo, rmax);
      float mnc = fmaxf(mn, NEG_BIG);
      float cf  = __expf(mo - mnc);
      float rs  = 0.0f;
#pragma unroll
      for (int t = 0; t < 4; ++t) {
        float pexp = __expf(st[t][r] - mnc);
        st[t][r] = pexp;
        rs += pexp;
      }
      lrow[r] = lrow[r] * cf + rs;
      mrow[r] = mn;
#pragma unroll
      for (int t = 0; t < 4; ++t) oacc[t][r] *= cf;
    }

    _Float16* sPw = &sP[wave * QW * LSTR];
#pragma unroll
    for (int t = 0; t < 4; ++t)
#pragma unroll
      for (int r = 0; r < 8; ++r)
        sPw[(r + 8 * grp) * LSTR + t * 16 + ln16] = (_Float16)st[t][r];
    __builtin_amdgcn_wave_barrier();
    const _Float16* prow = &sPw[ln16 * LSTR];
    V16U ap0, ap1;
    ap0.h[0] = *(const v8h*)(prow + grp * 8);
    ap0.h[1] = *(const v8h*)(prow + 16 + grp * 8);
    ap1.h[0] = *(const v8h*)(prow + 32 + grp * 8);
    ap1.h[1] = *(const v8h*)(prow + 48 + grp * 8);

#pragma unroll
    for (int t = 0; t < 4; ++t) {
      const _Float16* vrow = &sVt[(t * 16 + ln16) * LSTR];
      V16U bv0, bv1;
      bv0.h[0] = *(const v8h*)(vrow + grp * 8);
      bv0.h[1] = *(const v8h*)(vrow + 16 + grp * 8);
      bv1.h[0] = *(const v8h*)(vrow + 32 + grp * 8);
      bv1.h[1] = *(const v8h*)(vrow + 48 + grp * 8);
      oacc[t] = wmma_f16(ap0.v, bv0.v, oacc[t]);
      oacc[t] = wmma_f16(ap1.v, bv1.v, oacc[t]);
    }
  }

  float* obase = O + base;
#pragma unroll
  for (int r = 0; r < 8; ++r) {
    float ls = lrow[r];
#pragma unroll
    for (int off = 1; off < 16; off <<= 1)
      ls += __shfl_xor(ls, off, 32);
    float inv = 1.0f / ls;
    int m = q0 + r + 8 * grp;
    float* orow = obase + (size_t)m * DKK;
#pragma unroll
    for (int t = 0; t < 4; ++t)
      orow[t * 16 + ln16] = oacc[t][r] * inv;
  }
}

extern "C" void kernel_launch(void* const* d_in, const int* in_sizes, int n_in,
                              void* d_out, int out_size, void* d_ws, size_t ws_size,
                              hipStream_t stream) {
  const float* Q = (const float*)d_in[0];
  const float* K = (const float*)d_in[1];
  const float* V = (const float*)d_in[2];
  float* O = (float*)d_out;

  const size_t elems = (size_t)BHN * S_LEN * DKK;          // 4M
  const size_t need  = 2 * elems * sizeof(_Float16);       // Kh + Vt = 16 MiB
  const int qblocks = S_LEN / QB;
  dim3 grid(BHN * qblocks), block(256);

  if (ws_size >= need) {
    _Float16* Kh = (_Float16*)d_ws;
    _Float16* Vt = Kh + elems;
    cvt_kv<<<dim3((unsigned)(elems / 256)), 256, 0, stream>>>(K, V, Kh, Vt);
    fa_fwd_tdm<<<grid, block, 0, stream>>>(Q, Kh, Vt, O);
  } else {
    fa_fwd_manual<<<grid, block, 0, stream>>>(Q, K, V, O);
  }
}
